// GRUClassifier_7584912245265
// MI455X (gfx1250) — matrix operand order repaired and checked by reading, not compiled, not run on hardware
//
#include <hip/hip_runtime.h>

// ---------------- problem constants ----------------
#define BB 32
#define TT 4096
#define DX 128
#define HH 128
#define KK 10

#define WSTR 136    // f16 elements per padded weight/h/x row (128 + 8 pad)
#define HB   (16*WSTR) // halves per [16][WSTR] tile buffer
#define FCSTR 264   // f16 elements per padded W_fc row (256 + 8 pad)

// LDS: Wih+Whh (f16, padded) + biases (f32) + h dbl-buf + x dbl-buf
#define SCAN_LDS_BYTES (384*WSTR*2*2 + 384*4*2 + 16*WSTR*2*2 + 16*WSTR*2*2)

#define GAS __attribute__((address_space(1)))

typedef __attribute__((ext_vector_type(16))) _Float16 v16h;
typedef __attribute__((ext_vector_type(8)))  float    v8f;
typedef __attribute__((ext_vector_type(4)))  unsigned int u32x4;
typedef __attribute__((ext_vector_type(4)))  float        f32x4;
typedef __attribute__((ext_vector_type(8)))  int          i32x8;
typedef __attribute__((ext_vector_type(4)))  int          i32x4;

union FragU { u32x4 q[2]; v16h h; };

#define WMMA_F16(a,b,c) \
  __builtin_amdgcn_wmma_f32_16x16x32_f16(false,(a),false,(b),(short)0,(c),false,false)

// A-fragment (16x32 f16) from LDS: lane m = lid&15, khalf = lid>>4.
// Per-lane K: {0..7}+8*kh (16B at +0) and {16..23}+8*kh (16B at +32B).
__device__ __forceinline__ v16h load_fragA(const _Float16* base, int m, int strideH,
                                           int kk, int kh) {
  const _Float16* p = base + m*strideH + kk*32 + kh*8;
  FragU f;
  f.q[0] = *(const u32x4*)p;
  f.q[1] = *(const u32x4*)(p + 16);
  return f.h;
}

// A-fragment from GLOBAL memory (explicit AS(1) so we get global_load_b128).
__device__ __forceinline__ v16h load_fragA_g(const _Float16* base, int kk, int kh) {
  GAS const u32x4* p = (GAS const u32x4*)(size_t)(base + kk*32 + kh*8);
  FragU f;
  f.q[0] = p[0];
  f.q[1] = p[2];   // +32 bytes
  return f.h;
}

// B-fragment (32x16 f16) from LDS: lane n = lid&15, khalf = lid>>4.
// Per-lane K: {0..7}+8*kh (16B at +0) and {16..23}+8*kh (16B at +32B).
// W stored [out][in] row-major is exactly this layout.
__device__ __forceinline__ v16h load_fragB(const _Float16* base, int n, int strideH,
                                           int kk, int kh) {
  const _Float16* p = base + n*strideH + kk*32 + kh*8;
  FragU f;
  f.q[0] = *(const u32x4*)p;
  f.q[1] = *(const u32x4*)(p + 16);
  return f.h;
}

__device__ __forceinline__ float fast_sigmoid(float x) {
  float e = __builtin_amdgcn_exp2f(-1.4426950408889634f * x);
  return __builtin_amdgcn_rcpf(1.0f + e);
}
__device__ __forceinline__ float fast_tanh(float x) {
  float e = __builtin_amdgcn_exp2f(-2.8853900817779268f * x);
  return (1.0f - e) * __builtin_amdgcn_rcpf(1.0f + e);
}

// Low 32 bits of a generic LDS pointer = wave-relative LDS byte address.
__device__ __forceinline__ unsigned lds_off(const void* p) {
  return (unsigned)(unsigned long long)(size_t)p;
}

// ---------------------------------------------------------------------------
// TDM: DMA one [16 x 128] f16 tile (row stride TT*DX halves) into padded LDS
// rows of WSTR halves. D# per CDNA5 ISA.
//   group0: count=1 | lds_addr | global_addr | type=2
//   group1: data_size=1(2B), pad_enable, pad_interval=5 (64 DW = one 256B row),
//           pad_amount=3 (4 DW = 8 f16 -> row stride 136), dims/strides.
// This toolchain uses the 6-arg builtin: (g0, g1, g2, g3, g4, cpol).
// ---------------------------------------------------------------------------
__device__ __forceinline__ void tdm_load_x_tile(const _Float16* gsrc,
                                                unsigned lds_byte_off) {
  const unsigned long long ga = (unsigned long long)(size_t)gsrc;
  u32x4 g0;
  g0[0] = 1u;                                              // count=1 (valid)
  g0[1] = lds_byte_off;                                    // lds_addr
  g0[2] = (unsigned)ga;                                    // global_addr lo
  g0[3] = (unsigned)((ga >> 32) & 0x01FFFFFFull) | (2u << 30); // ga hi | type=2
  i32x8 g1;
  g1[0] = (1 << 16) | (1 << 20) | (5 << 22) | (3 << 25);   // dsz|pad_en|intvl|amt
  g1[1] = (int)(128u << 16);                               // tensor_dim0 = 128
  g1[2] = (int)(16u << 16);                                // tensor_dim1 = 16
  g1[3] = (int)(128u << 16);                               // tile_dim0 = 128
  g1[4] = 16;                                              // tile_dim1=16, dim2=0
  g1[5] = (int)(TT * DX);                                  // dim0_stride = 524288
  g1[6] = 0;
  g1[7] = 0;
  const i32x4 z4 = { 0, 0, 0, 0 };
  const i32x8 z8 = { 0, 0, 0, 0, 0, 0, 0, 0 };
  __builtin_amdgcn_tensor_load_to_lds(g0, g1, z4, z4, z8, 0);
}

// ---------------------------------------------------------------------------
// x pre-convert: f32 -> f16 once (bandwidth-bound, whole chip), so the scan's
// per-step critical path has no v_cvt chain. 8 floats -> one 16B store/thread.
// ---------------------------------------------------------------------------
extern "C" __global__ void __launch_bounds__(256)
x_cvt_kernel(const float* __restrict__ x, _Float16* __restrict__ x16) {
  const size_t i = ((size_t)blockIdx.x * 256 + threadIdx.x) * 8;
  GAS const f32x4* src = (GAS const f32x4*)(size_t)(x + i);
  const f32x4 a = src[0], b = src[1];
  union { _Float16 h[8]; u32x4 q; } o;
  o.h[0] = (_Float16)a.x; o.h[1] = (_Float16)a.y;
  o.h[2] = (_Float16)a.z; o.h[3] = (_Float16)a.w;
  o.h[4] = (_Float16)b.x; o.h[5] = (_Float16)b.y;
  o.h[6] = (_Float16)b.z; o.h[7] = (_Float16)b.w;
  *(GAS u32x4*)(size_t)(x16 + i) = o.q;
}

// ---------------------------------------------------------------------------
// Persistent GRU scan: grid = 4 workgroups (dir x batch-half), 256 threads.
// Wave w owns h-cols [16w,16w+16) across all 3 gates -> gate math in-register.
// Weights resident in LDS for all 4096 steps; x tiles DMA'd by the TDM.
// Double buffers selected by INTEGER offsets from one LDS base so all tile
// accesses stay provably AS(3) -> ds_load_b128 (no flat path in the loop).
// ---------------------------------------------------------------------------
extern "C" __global__ void __launch_bounds__(256, 1)
gru_scan_kernel(const _Float16* __restrict__ x16,
                const float* __restrict__ WihF, const float* __restrict__ WhhF,
                const float* __restrict__ bihF, const float* __restrict__ bhhF,
                const float* __restrict__ WihB, const float* __restrict__ WhhB,
                const float* __restrict__ bihB, const float* __restrict__ bhhB,
                _Float16* __restrict__ hsF, _Float16* __restrict__ hsB)
{
  extern __shared__ char smem[];
  _Float16* wihL = (_Float16*)smem;              // [384][WSTR]
  _Float16* whhL = wihL + 384 * WSTR;            // [384][WSTR]
  float*    bihL = (float*)(whhL + 384 * WSTR);
  float*    bhhL = bihL + 384;
  // tile area: h0 | h1 | x0 | x1, each [16][WSTR] f16, contiguous
  _Float16* tiles = (_Float16*)(bhhL + 384);

  const int dir   = blockIdx.x >> 1;
  const int bbase = (blockIdx.x & 1) * 16;
  const float* Wih = dir ? WihB : WihF;
  const float* Whh = dir ? WhhB : WhhF;
  const float* bih = dir ? bihB : bihF;
  const float* bhh = dir ? bhhB : bhhF;
  GAS _Float16* hs = (GAS _Float16*)(size_t)(dir ? hsB : hsF);

  // One-time: weights (f32 -> f16, padded rows), biases, h0 = 0.
  for (int idx = threadIdx.x; idx < 384 * 128; idx += 256) {
    const int r = idx >> 7, c = idx & 127;
    wihL[r * WSTR + c] = (_Float16)Wih[idx];
    whhL[r * WSTR + c] = (_Float16)Whh[idx];
  }
  for (int idx = threadIdx.x; idx < 384; idx += 256) {
    bihL[idx] = bih[idx];
    bhhL[idx] = bhh[idx];
  }
  for (int idx = threadIdx.x; idx < HB; idx += 256)
    tiles[idx] = (_Float16)0.0f;                 // h buffer 0

  const unsigned xoff0 = lds_off(tiles + 2 * HB);
  const int wave = threadIdx.x >> 5;

  // Prologue: DMA x tile for t=0 into x buffer 0.
  if (wave == 0) {
    const int tp0 = dir ? (TT - 1) : 0;
    tdm_load_x_tile(x16 + ((size_t)bbase * TT + (size_t)tp0) * DX, xoff0);
    __builtin_amdgcn_s_wait_tensorcnt(0);
  }
  __syncthreads();

  const int lid  = threadIdx.x & 31;
  const int lrow = lid & 15;
  const int kh   = lid >> 4;
  const int col  = wave * 16 + lrow;   // this lane's C-matrix column (h index)

  const float b_ir = bihL[col],       b_hr = bhhL[col];
  const float b_iz = bihL[128 + col], b_hz = bhhL[128 + col];
  const float b_in = bihL[256 + col], b_hn = bhhL[256 + col];

  for (int t = 0; t < TT; ++t) {
    const int cur = t & 1, nxt = cur ^ 1;
    const int hcuro = cur * HB;          // h tile offsets (halves)
    const int hnxto = nxt * HB;
    const int xcuro = (2 + cur) * HB;    // x tile offsets (halves)

    // Kick off next step's x tile DMA (overlaps the whole step below).
    if (wave == 0 && t + 1 < TT) {
      const int tp = dir ? (TT - 2 - t) : (t + 1);
      tdm_load_x_tile(x16 + ((size_t)bbase * TT + (size_t)tp) * DX,
                      xoff0 + (unsigned)(nxt * (HB * 2)));
    }

    v8f accx[3] = {};   // x-projection accumulators (r, z, n)
    v8f acch[3] = {};   // h-projection accumulators (r, z, n)
#pragma unroll
    for (int kk = 0; kk < 4; ++kk) {
      v16h ah = load_fragA(tiles + hcuro, lrow, WSTR, kk, kh);
      v16h ax = load_fragA(tiles + xcuro, lrow, WSTR, kk, kh);
#pragma unroll
      for (int g = 0; g < 3; ++g) {
        v16h bw = load_fragB(whhL, g * 128 + col, WSTR, kk, kh);
        acch[g] = WMMA_F16(ah, bw, acch[g]);
        v16h bi = load_fragB(wihL, g * 128 + col, WSTR, kk, kh);
        accx[g] = WMMA_F16(ax, bi, accx[g]);
      }
    }

    const int tphys = dir ? (TT - 1 - t) : t;
#pragma unroll
    for (int v = 0; v < 8; ++v) {
      const int m = v + 8 * kh;   // C layout: VGPR v, lanes 16-31 -> M = v+8
      const float rg = fast_sigmoid(accx[0][v] + b_ir + acch[0][v] + b_hr);
      const float zg = fast_sigmoid(accx[1][v] + b_iz + acch[1][v] + b_hz);
      const float ng = fast_tanh(accx[2][v] + b_in + rg * (acch[2][v] + b_hn));
      const float hold = (float)tiles[hcuro + m * WSTR + col];
      const float hnew = (1.0f - zg) * ng + zg * hold;
      tiles[hnxto + m * WSTR + col] = (_Float16)hnew;
      hs[((size_t)(bbase + m) * TT + (size_t)tphys) * HH + col] = (_Float16)hnew;
    }

    // TDM for step t+1 must have landed before anyone reads xs[nxt].
    if (wave == 0) __builtin_amdgcn_s_wait_tensorcnt(0);
    __syncthreads();
  }
}

// ---------------------------------------------------------------------------
// Classifier: out[b,t,k] = [hf|hb] @ W_fc^T + b_fc  (K padded 10 -> 16).
// 131072 rows -> 8192 M-tiles, 8 tiles per 256-thread block, 8 WMMAs/wave.
// ---------------------------------------------------------------------------
extern "C" __global__ void __launch_bounds__(256, 2)
gru_fc_kernel(const _Float16* __restrict__ hsF, const _Float16* __restrict__ hsB,
              const float* __restrict__ Wfc, const float* __restrict__ bfc,
              float* __restrict__ out)
{
  __shared__ _Float16 wfcL[16 * FCSTR];
  __shared__ float    bfcL[16];
  for (int idx = threadIdx.x; idx < 16 * 256; idx += 256) {
    const int r = idx >> 8, c = idx & 255;
    wfcL[r * FCSTR + c] = (_Float16)(r < KK ? Wfc[r * 256 + c] : 0.0f);
  }
  if (threadIdx.x < 16)
    bfcL[threadIdx.x] = (threadIdx.x < KK) ? bfc[threadIdx.x] : 0.0f;
  __syncthreads();

  const int wv   = threadIdx.x >> 5;
  const int lid  = threadIdx.x & 31;
  const int lrow = lid & 15;
  const int kh   = lid >> 4;

  const size_t r0 = ((size_t)blockIdx.x * 8 + (size_t)wv) * 16;
  const size_t r  = r0 + (size_t)lrow;              // row = b*T + t
  const _Float16* hfRow = hsF + r * HH;
  const _Float16* hbRow = hsB + r * HH;

  v8f acc = {};
#pragma unroll
  for (int kk = 0; kk < 4; ++kk) {                  // K = 0..127 (forward h)
    v16h a = load_fragA_g(hfRow, kk, kh);
    v16h b = load_fragB(wfcL, lrow, FCSTR, kk, kh);
    acc = WMMA_F16(a, b, acc);
  }
#pragma unroll
  for (int kk = 0; kk < 4; ++kk) {                  // K = 128..255 (backward h)
    v16h a = load_fragA_g(hbRow, kk, kh);
    v16h b = load_fragB(wfcL, lrow, FCSTR, kk + 4, kh);
    acc = WMMA_F16(a, b, acc);
  }

  if (lrow < KK) {                                  // C col n = lid&15; drop pad
    const float bv = bfcL[lrow];
    GAS float* og = (GAS float*)(size_t)out;
#pragma unroll
    for (int v = 0; v < 8; ++v) {
      const int m = v + 8 * kh;
      og[(r0 + (size_t)m) * KK + lrow] = acc[v] + bv;
    }
  }
}

// ---------------------------------------------------------------------------
extern "C" void kernel_launch(void* const* d_in, const int* in_sizes, int n_in,
                              void* d_out, int out_size, void* d_ws, size_t ws_size,
                              hipStream_t stream) {
  (void)in_sizes; (void)n_in; (void)out_size; (void)ws_size;
  const float* x    = (const float*)d_in[0];
  const float* WihF = (const float*)d_in[1];
  const float* WhhF = (const float*)d_in[2];
  const float* bihF = (const float*)d_in[3];
  const float* bhhF = (const float*)d_in[4];
  const float* WihB = (const float*)d_in[5];
  const float* WhhB = (const float*)d_in[6];
  const float* bihB = (const float*)d_in[7];
  const float* bhhB = (const float*)d_in[8];
  const float* Wfc  = (const float*)d_in[9];
  const float* bfc  = (const float*)d_in[10];

  // Workspace: hf, hb, x16 -- each [B][T][128] f16 (32 MiB each).
  _Float16* hsF = (_Float16*)d_ws;
  _Float16* hsB = hsF + (size_t)BB * TT * HH;
  _Float16* x16 = hsB + (size_t)BB * TT * HH;

  x_cvt_kernel<<<dim3((BB * TT * DX) / (8 * 256)), dim3(256), 0, stream>>>(x, x16);

  (void)hipFuncSetAttribute(reinterpret_cast<const void*>(gru_scan_kernel),
                            hipFuncAttributeMaxDynamicSharedMemorySize,
                            SCAN_LDS_BYTES);
  gru_scan_kernel<<<dim3(4), dim3(256), SCAN_LDS_BYTES, stream>>>(
      x16, WihF, WhhF, bihF, bhhF, WihB, WhhB, bihB, bhhB, hsF, hsB);

  gru_fc_kernel<<<dim3((BB * TT) / (16 * 8)), dim3(256), 0, stream>>>(
      hsF, hsB, Wfc, bfc, (float*)d_out);
}
